// SharedFullBandEncoder_8899172237593
// MI455X (gfx1250) — matrix operand order repaired and checked by reading, not compiled, not run on hardware
//
#include <hip/hip_runtime.h>

#define NN 100000        // nodes
#define NE 1600000       // edges (without self loops)
#define ET 1700000       // edges + self loops
#define ROWTILES (NN/16) // 6250 (exact)

typedef __attribute__((ext_vector_type(16))) __bf16 v16bf;
typedef __attribute__((ext_vector_type(8)))  float v8f;

// ---------- GEMM: H[NN x DOUT] = X[NN x DIN] @ W[DIN x DOUT], bf16 WMMA ----------
// Whole W staged in LDS (bf16, WMMA-B layout); each wave keeps its A row-tile in
// registers across all column tiles -> x is read exactly once from HBM.
template<int DIN, int DOUT>
__global__ void gat_gemm(const float* __restrict__ X, const float* __restrict__ W,
                         float* __restrict__ H) {
    constexpr int KSTEPS = DIN / 32;
    constexpr int CTILES = DOUT / 16;
    __shared__ __align__(32) __bf16 Wlds[CTILES][KSTEPS][32][16];

    // Stage W: lane (n = lane&15, half = lane>>4) element i of tile (ct, ks)
    // holds W[ks*32 + (i<8 ? half*8+i : 16+half*8+i-8)][ct*16 + n]  (B-matrix register layout)
    for (int t = threadIdx.x; t < CTILES * KSTEPS * 512; t += 256) {
        int i    = t & 15;
        int lane = (t >> 4) & 31;
        int ks   = (t >> 9) % KSTEPS;
        int ct   = t / (KSTEPS * 512);
        int n    = lane & 15;
        int half = lane >> 4;
        int k    = ks * 32 + (i < 8 ? half * 8 + i : 16 + half * 8 + (i - 8));
        Wlds[ct][ks][lane][i] = (__bf16)W[k * DOUT + ct * 16 + n];
    }
    __syncthreads();

    const int lane = threadIdx.x & 31;
    const int wid  = threadIdx.x >> 5;
    const int m    = lane & 15;
    const int half = lane >> 4;

    for (int rt = blockIdx.x * 8 + wid; rt < ROWTILES; rt += gridDim.x * 8) {
        const float* xr = X + (long long)(rt * 16 + m) * DIN;

        // Load A row-tile for all K once (fp32 -> bf16, packed cvt)
        v16bf areg[KSTEPS];
#pragma unroll
        for (int ks = 0; ks < KSTEPS; ks++) {
            // A layout: lane(m,half) holds K = half*8+0..7 and 16+half*8+0..7
            int k0 = ks * 32 + half * 8;
            float4 f0 = *(const float4*)(xr + k0);
            float4 f1 = *(const float4*)(xr + k0 + 4);
            float4 f2 = *(const float4*)(xr + k0 + 16);
            float4 f3 = *(const float4*)(xr + k0 + 20);
            v16bf a;
            a[0]=(__bf16)f0.x;  a[1]=(__bf16)f0.y;  a[2]=(__bf16)f0.z;  a[3]=(__bf16)f0.w;
            a[4]=(__bf16)f1.x;  a[5]=(__bf16)f1.y;  a[6]=(__bf16)f1.z;  a[7]=(__bf16)f1.w;
            a[8]=(__bf16)f2.x;  a[9]=(__bf16)f2.y;  a[10]=(__bf16)f2.z; a[11]=(__bf16)f2.w;
            a[12]=(__bf16)f3.x; a[13]=(__bf16)f3.y; a[14]=(__bf16)f3.z; a[15]=(__bf16)f3.w;
            areg[ks] = a;
        }

        // Sweep all column tiles, reusing A registers
#pragma unroll
        for (int ct = 0; ct < CTILES; ct++) {
            v8f c = {};
#pragma unroll
            for (int ks = 0; ks < KSTEPS; ks++) {
                v16bf b = *(const v16bf*)&Wlds[ct][ks][lane][0];
                c = __builtin_amdgcn_wmma_f32_16x16x32_bf16(
                        false, areg[ks], false, b, (short)0, c, false, false);
            }
            // D layout: lane n = lane&15; VGPR r -> row (half*8 + r)
            float* hr = H + (long long)(rt * 16 + half * 8) * DOUT + ct * 16 + m;
#pragma unroll
            for (int r = 0; r < 8; r++) hr[r * DOUT] = c[r];
        }
    }
}

// ---------- per-node attention scores: es = h . a_src, ed = h . a_dst ----------
__global__ void node_scores(const float* __restrict__ h, const float* __restrict__ a_s,
                            const float* __restrict__ a_d, float* es, float* ed, int F) {
    int w    = (blockIdx.x * blockDim.x + threadIdx.x) >> 5;
    int lane = threadIdx.x & 31;
    if (w >= NN) return;
    const float* hr = h + (long long)w * F;
    float s = 0.f, d = 0.f;
    for (int c = lane; c < F; c += 32) {
        float v = hr[c];
        s += v * a_s[c];
        d += v * a_d[c];
    }
    for (int off = 16; off; off >>= 1) {
        s += __shfl_down(s, off);
        d += __shfl_down(d, off);
    }
    if (lane == 0) { es[w] = s; ed[w] = d; }
}

// ---------- zero fill ----------
__global__ void zero_f32(float* p, int n) {
    int i = blockIdx.x * blockDim.x + threadIdx.x;
    if (i < n) p[i] = 0.f;
}

// ---------- edge pass 1: leaky-relu logit + segment max (monotone uint encoding) ----
__global__ void edge_logits(const long long* __restrict__ ei, const float* __restrict__ es,
                            const float* __restrict__ ed, float* e_val, unsigned* menc) {
    int i = blockIdx.x * blockDim.x + threadIdx.x;
    if (i >= ET) return;
    int s, d;
    if (i < NE) { s = (int)ei[i]; d = (int)ei[NE + i]; }
    else        { s = d = i - NE; }
    float e = es[s] + ed[d];
    e = (e > 0.f) ? e : 0.2f * e;
    e_val[i] = e;
    unsigned u = __float_as_uint(e);
    u = (u & 0x80000000u) ? ~u : (u | 0x80000000u);
    atomicMax(&menc[d], u);
}

// ---------- edge pass 2: exp(e - max) + segment sum ----------
__global__ void edge_expsum(const long long* __restrict__ ei, const float* __restrict__ e_val,
                            const unsigned* __restrict__ menc, float* denom, float* ex) {
    int i = blockIdx.x * blockDim.x + threadIdx.x;
    if (i >= ET) return;
    int d = (i < NE) ? (int)ei[NE + i] : (i - NE);
    unsigned u = menc[d];
    u = (u & 0x80000000u) ? (u & 0x7FFFFFFFu) : ~u;
    float m = __uint_as_float(u);
    float v = __expf(e_val[i] - m);
    ex[i] = v;
    atomicAdd(&denom[d], v);
}

// ---------- edge pass 3: agg[dst] += alpha * h[src], one wave per edge ----------
template<int F>
__global__ void edge_aggregate(const long long* __restrict__ ei, const float* __restrict__ ex,
                               const float* __restrict__ denom, const float* __restrict__ h,
                               float* agg) {
    int w    = (blockIdx.x * blockDim.x + threadIdx.x) >> 5;
    int lane = threadIdx.x & 31;
    if (w >= ET) return;
    int s, d;
    if (w < NE) { s = (int)ei[w]; d = (int)ei[NE + w]; }
    else        { s = d = w - NE; }
    float alpha = ex[w] / denom[d];
    const float* hs = h + (long long)s * F;
    float* od = agg + (long long)d * F;
    constexpr int PER = F / 32;
    int base = lane * PER;
    float vals[PER];
#pragma unroll
    for (int j = 0; j < PER; j++) vals[j] = hs[base + j];
#pragma unroll
    for (int j = 0; j < PER; j++) atomicAdd(&od[base + j], alpha * vals[j]);
}

// ---------- epilogue: y = relu((agg + b) * gamma/sqrt(1+eps) + beta) ----------
__global__ void bias_bn_relu(const float* __restrict__ agg, const float* __restrict__ b,
                             const float* __restrict__ g, const float* __restrict__ be,
                             float* y, int F, int total) {
    int i = blockIdx.x * blockDim.x + threadIdx.x;
    if (i >= total) return;
    int c = i & (F - 1);  // F is a power of two
    float scale = g[c] * rsqrtf(1.0f + 1e-5f);
    float v = (agg[i] + b[c]) * scale + be[c];
    y[i] = (v > 0.f) ? v : 0.f;
}

// ---------- layer driver ----------
template<int DIN, int DOUT>
static void run_layer(const float* x, const long long* ei,
                      const float* W, const float* as, const float* ad,
                      const float* b, const float* g, const float* be,
                      float* h, float* agg_y,
                      float* e_src, float* e_dst, unsigned* menc, float* denom,
                      float* e_val, float* exb, hipStream_t stream) {
    gat_gemm<DIN, DOUT><<<782, 256, 0, stream>>>(x, W, h);
    node_scores<<<(NN * 32) / 256, 256, 0, stream>>>(h, as, ad, e_src, e_dst, DOUT);
    zero_f32<<<(NN * DOUT + 255) / 256, 256, 0, stream>>>(agg_y, NN * DOUT);
    zero_f32<<<(NN + 255) / 256, 256, 0, stream>>>((float*)menc, NN);
    zero_f32<<<(NN + 255) / 256, 256, 0, stream>>>(denom, NN);
    edge_logits<<<(ET + 255) / 256, 256, 0, stream>>>(ei, e_src, e_dst, e_val, menc);
    edge_expsum<<<(ET + 255) / 256, 256, 0, stream>>>(ei, e_val, menc, denom, exb);
    edge_aggregate<DOUT><<<ET / 8, 256, 0, stream>>>(ei, exb, denom, h, agg_y);
    bias_bn_relu<<<(NN * DOUT + 255) / 256, 256, 0, stream>>>(agg_y, b, g, be, agg_y,
                                                              DOUT, NN * DOUT);
}

extern "C" void kernel_launch(void* const* d_in, const int* in_sizes, int n_in,
                              void* d_out, int out_size, void* d_ws, size_t ws_size,
                              hipStream_t stream) {
    const float*     x  = (const float*)d_in[0];
    const long long* ei = (const long long*)d_in[1];  // int64 (2, NE) row-major

    // per-layer params: W, a_src, a_dst, b, gamma, beta at 2.., 8.., 14..
    const float* W1 = (const float*)d_in[2];  const float* as1 = (const float*)d_in[3];
    const float* ad1 = (const float*)d_in[4]; const float* b1  = (const float*)d_in[5];
    const float* g1  = (const float*)d_in[6]; const float* be1 = (const float*)d_in[7];
    const float* W2 = (const float*)d_in[8];  const float* as2 = (const float*)d_in[9];
    const float* ad2 = (const float*)d_in[10]; const float* b2 = (const float*)d_in[11];
    const float* g2  = (const float*)d_in[12]; const float* be2 = (const float*)d_in[13];
    const float* W3 = (const float*)d_in[14]; const float* as3 = (const float*)d_in[15];
    const float* ad3 = (const float*)d_in[16]; const float* b3 = (const float*)d_in[17];
    const float* g3  = (const float*)d_in[18]; const float* be3 = (const float*)d_in[19];

    // workspace layout (bytes)
    char* ws = (char*)d_ws;
    float*    bufA  = (float*)(ws);                    // NN*256 f32 (h scratch)
    float*    bufB  = (float*)(ws + 102400000);        // NN*256 f32 (y1)
    float*    bufC  = (float*)(ws + 204800000);        // NN*128 f32 (y2)
    float*    e_src = (float*)(ws + 256000000);        // NN
    float*    e_dst = e_src + NN;                      // NN
    unsigned* menc  = (unsigned*)(e_dst + NN);         // NN
    float*    denom = (float*)(menc + NN);             // NN
    float*    e_val = denom + NN;                      // ET
    float*    exb   = e_val + ET;                      // ET

    // Layer 1: 128 -> 256 : x -> h(bufA) -> y1(bufB)
    run_layer<128, 256>(x, ei, W1, as1, ad1, b1, g1, be1,
                        bufA, bufB, e_src, e_dst, menc, denom, e_val, exb, stream);
    // Layer 2: 256 -> 128 : y1(bufB) -> h(bufA) -> y2(bufC)
    run_layer<256, 128>(bufB, ei, W2, as2, ad2, b2, g2, be2,
                        bufA, bufC, e_src, e_dst, menc, denom, e_val, exb, stream);
    // Layer 3: 128 -> 64 : y2(bufC) -> h(bufA) -> d_out
    run_layer<128, 64>(bufC, ei, W3, as3, ad3, b3, g3, be3,
                       bufA, (float*)d_out, e_src, e_dst, menc, denom, e_val, exb, stream);
}
